// GRUcell_83562883711201
// MI455X (gfx1250) — matrix operand order repaired and checked by reading, not compiled, not run on hardware
//
#include <hip/hip_runtime.h>
#include <hip/hip_bf16.h>

// ---------------- problem constants ----------------
#define Bdim 64
#define Tdim 512
#define Cdim 1024
#define HROW 1032   // LDS row stride in ushorts (1024 + 8 pad -> conflict-reduced b128 reads)

typedef __attribute__((ext_vector_type(16))) __bf16 v16bf;
typedef __attribute__((ext_vector_type(8)))  float  v8f;

union Frag16 { v16bf v; uint4 q[2]; };

// ---- bf16 <-> f32 helpers (round-to-nearest-even) ----
static __device__ __forceinline__ unsigned short f2bf(float x) {
    unsigned u = __float_as_uint(x);
    return (unsigned short)((u + 0x7FFFu + ((u >> 16) & 1u)) >> 16);
}
static __device__ __forceinline__ float bf2f(unsigned short b) {
    return __uint_as_float(((unsigned)b) << 16);
}

// A fragment (16x32 bf16, MxK): lane holds row m = ml; elements 0..7 = K[k0+8h .. +7],
// elements 8..15 = K[k0+8h+16 .. +23]   (ISA 05_wmma 7.12.2, 16-bit A 16x32)
static __device__ __forceinline__ void load_fa(Frag16& f, const unsigned short* row, int k0, int h) {
    f.q[0] = *reinterpret_cast<const uint4*>(row + k0 + 8 * h);
    f.q[1] = *reinterpret_cast<const uint4*>(row + k0 + 8 * h + 16);
}
// B fragment (32x16 bf16, KxN): lane holds col n = ml; elements 0..7 = K[k0+8h .. +7],
// elements 8..15 = K[k0+8h+16 .. +23]
static __device__ __forceinline__ void load_fb(Frag16& f, const unsigned short* row, int k0, int h) {
    f.q[0] = *reinterpret_cast<const uint4*>(row + k0 + 8 * h);
    f.q[1] = *reinterpret_cast<const uint4*>(row + k0 + 8 * h + 16);
}

static __device__ __forceinline__ v8f wmma_bf16(const Frag16& a, const Frag16& b, v8f c) {
    return __builtin_amdgcn_wmma_f32_16x16x32_bf16(false, a.v, false, b.v, (short)0, c,
                                                   false, false);
}

// Cooperative copy of a [64 x 1024] bf16 matrix (global, contiguous) into LDS with
// padded rows (HROW).  256 threads x 32 iterations x 16B = 128 KB.
static __device__ __forceinline__ void stage_to_lds(unsigned short* lds,
                                                    const unsigned short* __restrict__ g) {
    const uint4* src = reinterpret_cast<const uint4*>(g);
    const int tid = threadIdx.x;
#pragma unroll
    for (int i = 0; i < 32; ++i) {
        int idx = i * 256 + tid;          // 8192 chunks of 8 ushorts
        int row = idx >> 7;               // chunk / 128
        int col = (idx & 127) * 8;
        *reinterpret_cast<uint4*>(lds + row * HROW + col) = src[idx];
    }
}

// ---------------- conversion kernels ----------------
__global__ void k_cvt_bf16(const float* __restrict__ s, unsigned short* __restrict__ d, int n4) {
    int i = blockIdx.x * blockDim.x + threadIdx.x;
    if (i >= n4) return;
    float4 v = reinterpret_cast<const float4*>(s)[i];
    unsigned lo = (unsigned)f2bf(v.x) | ((unsigned)f2bf(v.y) << 16);
    unsigned hi = (unsigned)f2bf(v.z) | ((unsigned)f2bf(v.w) << 16);
    reinterpret_cast<uint2*>(d)[i] = make_uint2(lo, hi);
}

__global__ void k_init_h(const float* __restrict__ h0, float* __restrict__ Hf,
                         unsigned short* __restrict__ Hb, int n) {
    int i = blockIdx.x * blockDim.x + threadIdx.x;
    if (i >= n) return;
    float v = h0[i];
    Hf[i] = v;
    Hb[i] = f2bf(v);
}

// ---------------- big NT GEMM: O[m,n] = A[m,:] . W[n,:] + bias[n] ----------------
// 8 waves/WG as 4(m) x 2(n); each wave owns a 32x64 tile (2x4 WMMA tiles) -> WG = 128x128.
// mode 0: Obf[m*C+n] = bf16(val)               (X projections, bias = bx*)
// mode 1: Of32[b*T*C + t*C + n] = val, m=t*B+b (final Y projection into d_out)
static __device__ __forceinline__ void stile(const v8f& c, int mbase, int nbase, int h, int ml,
                                             const float* __restrict__ bias,
                                             unsigned short* __restrict__ Obf,
                                             float* __restrict__ Of32, int mode) {
    int n = nbase + ml;
    float bv = bias[n];
#pragma unroll
    for (int j = 0; j < 8; ++j) {
        int m = mbase + j + 8 * h;
        float v = c[j] + bv;
        if (mode == 0) {
            Obf[(size_t)m * Cdim + n] = f2bf(v);
        } else {
            int t = m >> 6;       // m / Bdim
            int b = m & 63;       // m % Bdim
            Of32[(size_t)b * (Tdim * Cdim) + (size_t)t * Cdim + n] = v;
        }
    }
}

__global__ __launch_bounds__(256) void k_gemm_bf16(const unsigned short* __restrict__ A,
                                                   const unsigned short* __restrict__ W,
                                                   const float* __restrict__ bias,
                                                   unsigned short* __restrict__ Obf,
                                                   float* __restrict__ Of32, int mode) {
    const int lane = threadIdx.x & 31;
    const int w    = threadIdx.x >> 5;
    const int wm   = w & 3, wn = w >> 2;             // 4x2 waves
    const int m0   = blockIdx.x * 128 + wm * 32;
    const int n0   = blockIdx.y * 128 + wn * 64;
    const int h    = lane >> 4, ml = lane & 15;

    const unsigned short* a0p = A + (size_t)(m0 + ml) * Cdim;
    const unsigned short* a1p = a0p + 16 * Cdim;
    const unsigned short* bp[4];
#pragma unroll
    for (int nt = 0; nt < 4; ++nt) bp[nt] = W + (size_t)(n0 + nt * 16 + ml) * Cdim;

    v8f c0[4] = {}, c1[4] = {};
    for (int k0 = 0; k0 < Cdim; k0 += 32) {
        Frag16 a0, a1, b[4];
        load_fa(a0, a0p, k0, h);
        load_fa(a1, a1p, k0, h);
#pragma unroll
        for (int nt = 0; nt < 4; ++nt) load_fb(b[nt], bp[nt], k0, h);
#pragma unroll
        for (int nt = 0; nt < 4; ++nt) {
            c0[nt] = wmma_bf16(a0, b[nt], c0[nt]);
            c1[nt] = wmma_bf16(a1, b[nt], c1[nt]);
        }
    }
#pragma unroll
    for (int nt = 0; nt < 4; ++nt) {
        stile(c0[nt], m0,      n0 + nt * 16, h, ml, bias, Obf, Of32, mode);
        stile(c1[nt], m0 + 16, n0 + nt * 16, h, ml, bias, Obf, Of32, mode);
    }
}

// ---------------- per-step kernel 1: Z, R gates (fused, LDS-staged H) --------------
// Z = sigmoid(Xz[t] + H.Whz^T + bhz); R = sigmoid(Xr[t] + H.Whr^T + bhr); RH = bf16(R*H)
__global__ __launch_bounds__(256) void k_gates(const unsigned short* __restrict__ Hb,
                                               const float* __restrict__ Hf,
                                               const unsigned short* __restrict__ Wz,
                                               const unsigned short* __restrict__ Wr,
                                               const float* __restrict__ bz,
                                               const float* __restrict__ br,
                                               const unsigned short* __restrict__ Xz,
                                               const unsigned short* __restrict__ Xr,
                                               float* __restrict__ Zf,
                                               unsigned short* __restrict__ RHb, int t) {
    extern __shared__ unsigned short ldsH[];         // 64 x HROW
    stage_to_lds(ldsH, Hb);
    __syncthreads();

    const int lane  = threadIdx.x & 31;
    const int w     = threadIdx.x >> 5;
    const int ntile = blockIdx.x * 8 + w;            // 0..63
    const int n0    = ntile * 16;
    const int h     = lane >> 4, ml = lane & 15;

    const unsigned short* ap[4];
#pragma unroll
    for (int mt = 0; mt < 4; ++mt) ap[mt] = ldsH + (16 * mt + ml) * HROW;
    const unsigned short* bzp = Wz + (size_t)(n0 + ml) * Cdim;
    const unsigned short* brp = Wr + (size_t)(n0 + ml) * Cdim;

    v8f cz[4] = {}, cr[4] = {};
    for (int k0 = 0; k0 < Cdim; k0 += 32) {
        Frag16 a[4], fz, fr;
#pragma unroll
        for (int mt = 0; mt < 4; ++mt) load_fa(a[mt], ap[mt], k0, h);
        load_fb(fz, bzp, k0, h);
        load_fb(fr, brp, k0, h);
#pragma unroll
        for (int mt = 0; mt < 4; ++mt) {
            cz[mt] = wmma_bf16(a[mt], fz, cz[mt]);
            cr[mt] = wmma_bf16(a[mt], fr, cr[mt]);
        }
    }
    const int n = n0 + ml;
    const float bzv = bz[n], brv = br[n];
#pragma unroll
    for (int mt = 0; mt < 4; ++mt) {
#pragma unroll
        for (int j = 0; j < 8; ++j) {
            int bb = 16 * mt + j + 8 * h;            // batch row
            size_t xidx = (size_t)bb * (Tdim * Cdim) + (size_t)t * Cdim + n;
            size_t hidx = (size_t)bb * Cdim + n;
            float zp = cz[mt][j] + bf2f(Xz[xidx]) + bzv;
            float rp = cr[mt][j] + bf2f(Xr[xidx]) + brv;
            float z = 1.0f / (1.0f + __expf(-zp));
            float r = 1.0f / (1.0f + __expf(-rp));
            float hv = Hf[hidx];
            Zf[hidx]  = z;
            RHb[hidx] = f2bf(r * hv);
        }
    }
}

// ---------------- per-step kernel 2: candidate + state update (LDS-staged RH) ----------
// Ht = tanh(Xh[t] + RH.Whh^T + bhh); Hn = Z*H + (1-Z)*Ht; store Hf/Hb and Hs[t] (bf16)
__global__ __launch_bounds__(256) void k_cand(const unsigned short* __restrict__ RHb,
                                              const unsigned short* __restrict__ Wh,
                                              const float* __restrict__ bh,
                                              const unsigned short* __restrict__ Xh,
                                              const float* __restrict__ Zf,
                                              float* __restrict__ Hf,
                                              unsigned short* __restrict__ Hb,
                                              unsigned short* __restrict__ Hs, int t) {
    extern __shared__ unsigned short ldsA[];         // 64 x HROW
    stage_to_lds(ldsA, RHb);
    __syncthreads();

    const int lane  = threadIdx.x & 31;
    const int w     = threadIdx.x >> 5;
    const int ntile = blockIdx.x * 8 + w;
    const int n0    = ntile * 16;
    const int h     = lane >> 4, ml = lane & 15;

    const unsigned short* ap[4];
#pragma unroll
    for (int mt = 0; mt < 4; ++mt) ap[mt] = ldsA + (16 * mt + ml) * HROW;
    const unsigned short* bp = Wh + (size_t)(n0 + ml) * Cdim;

    v8f ch[4] = {};
    for (int k0 = 0; k0 < Cdim; k0 += 32) {
        Frag16 a[4], fb;
#pragma unroll
        for (int mt = 0; mt < 4; ++mt) load_fa(a[mt], ap[mt], k0, h);
        load_fb(fb, bp, k0, h);
#pragma unroll
        for (int mt = 0; mt < 4; ++mt) ch[mt] = wmma_bf16(a[mt], fb, ch[mt]);
    }
    const int n = n0 + ml;
    const float bhv = bh[n];
    unsigned short* HsT = Hs + (size_t)t * (Bdim * Cdim);
#pragma unroll
    for (int mt = 0; mt < 4; ++mt) {
#pragma unroll
        for (int j = 0; j < 8; ++j) {
            int bb = 16 * mt + j + 8 * h;
            size_t xidx = (size_t)bb * (Tdim * Cdim) + (size_t)t * Cdim + n;
            size_t hidx = (size_t)bb * Cdim + n;
            float pre  = ch[mt][j] + bf2f(Xh[xidx]) + bhv;
            float htil = tanhf(pre);
            float z  = Zf[hidx];
            float hv = Hf[hidx];
            float hn = z * hv + (1.0f - z) * htil;
            Hf[hidx] = hn;
            unsigned short hbv = f2bf(hn);
            Hb[hidx]  = hbv;
            HsT[hidx] = hbv;                         // Hs layout [T,B,C]
        }
    }
}

// ---------------- host-side orchestration ----------------
extern "C" void kernel_launch(void* const* d_in, const int* in_sizes, int n_in,
                              void* d_out, int out_size, void* d_ws, size_t ws_size,
                              hipStream_t stream) {
    const float* X   = (const float*)d_in[0];
    const float* h0  = (const float*)d_in[1];
    const float* Wxz = (const float*)d_in[2];
    const float* bxz = (const float*)d_in[3];
    const float* Whz = (const float*)d_in[4];
    const float* bhz = (const float*)d_in[5];
    const float* Wxr = (const float*)d_in[6];
    const float* bxr = (const float*)d_in[7];
    const float* Whr = (const float*)d_in[8];
    const float* bhr = (const float*)d_in[9];
    const float* Wxh = (const float*)d_in[10];
    const float* bxh = (const float*)d_in[11];
    const float* Whh = (const float*)d_in[12];
    const float* bhh = (const float*)d_in[13];
    const float* Whq = (const float*)d_in[14];
    const float* bhq = (const float*)d_in[15];

    const size_t nBTC = (size_t)Bdim * Tdim * Cdim;   // 33,554,432
    const size_t nCC  = (size_t)Cdim * Cdim;          // 1,048,576
    const size_t nBC  = (size_t)Bdim * Cdim;          // 65,536

    char* wp = (char*)d_ws;
    auto walloc = [&](size_t bytes) -> char* {
        char* p = wp;
        wp += (bytes + 255) & ~(size_t)255;
        return p;
    };
    unsigned short* Xbf  = (unsigned short*)walloc(nBTC * 2);
    unsigned short* bWxz = (unsigned short*)walloc(nCC * 2);
    unsigned short* bWhz = (unsigned short*)walloc(nCC * 2);
    unsigned short* bWxr = (unsigned short*)walloc(nCC * 2);
    unsigned short* bWhr = (unsigned short*)walloc(nCC * 2);
    unsigned short* bWxh = (unsigned short*)walloc(nCC * 2);
    unsigned short* bWhh = (unsigned short*)walloc(nCC * 2);
    unsigned short* bWhq = (unsigned short*)walloc(nCC * 2);
    unsigned short* Xzb  = (unsigned short*)walloc(nBTC * 2);  // [B,T,C], bias folded
    unsigned short* Xrb  = (unsigned short*)walloc(nBTC * 2);
    unsigned short* Xhb  = (unsigned short*)walloc(nBTC * 2);
    unsigned short* Hs   = (unsigned short*)walloc(nBTC * 2);  // [T,B,C]
    float*          Hf   = (float*)walloc(nBC * 4);
    unsigned short* Hb   = (unsigned short*)walloc(nBC * 2);
    float*          Zf   = (float*)walloc(nBC * 4);
    unsigned short* RHb  = (unsigned short*)walloc(nBC * 2);

    // 1) convert inputs to bf16
    {
        int n4 = (int)(nBTC / 4);
        k_cvt_bf16<<<n4 / 256, 256, 0, stream>>>(X, Xbf, n4);
        n4 = (int)(nCC / 4);
        k_cvt_bf16<<<n4 / 256, 256, 0, stream>>>(Wxz, bWxz, n4);
        k_cvt_bf16<<<n4 / 256, 256, 0, stream>>>(Whz, bWhz, n4);
        k_cvt_bf16<<<n4 / 256, 256, 0, stream>>>(Wxr, bWxr, n4);
        k_cvt_bf16<<<n4 / 256, 256, 0, stream>>>(Whr, bWhr, n4);
        k_cvt_bf16<<<n4 / 256, 256, 0, stream>>>(Wxh, bWxh, n4);
        k_cvt_bf16<<<n4 / 256, 256, 0, stream>>>(Whh, bWhh, n4);
        k_cvt_bf16<<<n4 / 256, 256, 0, stream>>>(Whq, bWhq, n4);
        k_init_h<<<(int)(nBC / 256), 256, 0, stream>>>(h0, Hf, Hb, (int)nBC);
    }

    // 2) batched input projections: Xz/Xr/Xh = X.Wx*^T + bx* (bf16, [B,T,C] layout)
    dim3 gG(Bdim * Tdim / 128, Cdim / 128);  // (256, 8)
    k_gemm_bf16<<<gG, 256, 0, stream>>>(Xbf, bWxz, bxz, Xzb, nullptr, 0);
    k_gemm_bf16<<<gG, 256, 0, stream>>>(Xbf, bWxr, bxr, Xrb, nullptr, 0);
    k_gemm_bf16<<<gG, 256, 0, stream>>>(Xbf, bWxh, bxh, Xhb, nullptr, 0);

    // 3) sequential scan (Y projection hoisted out of the loop); 132 KB dynamic LDS/WG
    const size_t ldsBytes = (size_t)Bdim * HROW * 2;
    for (int t = 0; t < Tdim; ++t) {
        k_gates<<<8, 256, ldsBytes, stream>>>(Hb, Hf, bWhz, bWhr, bhz, bhr, Xzb, Xrb, Zf, RHb, t);
        k_cand<<<8, 256, ldsBytes, stream>>>(RHb, bWhh, bhh, Xhb, Zf, Hf, Hb, Hs, t);
    }

    // 4) Y = Hs.Whq^T + bhq, with [T,B,C] -> [B,T,C] remap in the epilogue
    k_gemm_bf16<<<gG, 256, 0, stream>>>(Hs, bWhq, bhq, nullptr, (float*)d_out, 1);
}
